// sccnn_conv_no_n_to_e_8976481648965
// MI455X (gfx1250) — matrix operand order repaired and checked by reading, compiled, not run on hardware
//
#include <hip/hip_runtime.h>

// ---------------------------------------------------------------------------
// CDNA5 / gfx1250 SCCNN forward pass.
// Heavy math via v_wmma_f32_16x16x32_bf16 (f32 accumulate).
// - B operands pre-converted once to bf16, transposed [128][K].
// - Independent Chebyshev chains sharing a Laplacian fused via gridDim.y=2.
// - Chebyshev pass 1 writes a bf16 copy of the Laplacian (fits L2);
//   pass 2 reads it back: half the bytes, mostly L2 hits, no convert VALU.
// ---------------------------------------------------------------------------

typedef __attribute__((ext_vector_type(16))) __bf16 v16bf;
typedef __attribute__((ext_vector_type(8)))  float  v8f;

#define TILE_M 64
#define TILE_K 32
#define BLK    128            // 4 waves (wave32)
#define LDSA_STRIDE 80        // 32 bf16 = 64B payload + 16B pad (bank rotation)
#define LDSB_STRIDE 80

#define N0 2048
#define N1 8192
#define N2 4096

// AMODE: 0 = A is f32 (convert on the fly)
//        1 = A is f32; additionally write packed bf16 copy to Aout (row-major)
//        2 = A is bf16 (read directly, no conversion)
#define AM_F32   0
#define AM_WRITE 1
#define AM_BF16  2

__device__ __forceinline__ unsigned int f2bf_pair(float lo, float hi) {
  union { __bf16 h[2]; unsigned int u; } cv;
  cv.h[0] = (__bf16)lo;
  cv.h[1] = (__bf16)hi;
  return cv.u;
}
__device__ __forceinline__ unsigned short f2bf1(float x) {
  union { __bf16 h; unsigned short u; } cv;
  cv.h = (__bf16)x;
  return cv.u;
}

union FragCvt { uint4 u[2]; v16bf v; };

// C[M,128] = op(A) @ Bt[bf16,col-major] (+ epilogue variants).
// K % 32 == 0, M % 64 == 0. Bt is [128 cols][K rows] bf16, row stride ldBt.
// blockIdx.y selects (Bt0,C0,P0,ldP0) or (Bt1,C1,P1,ldP1) -> fused dual-chain.
template<int AMODE, bool TRANSA, bool CHEB, bool RELU, bool ROWSCALE>
__global__ __launch_bounds__(BLK)
void gemm_bf16_wmma(const void* __restrict__ Aop, int ldA,
                    unsigned short* __restrict__ Aout,
                    const unsigned short* __restrict__ Bt0,
                    const unsigned short* __restrict__ Bt1, int ldBt,
                    float* __restrict__ C0, float* __restrict__ C1, int ldC,
                    const float* __restrict__ P0, const float* __restrict__ P1,
                    int ldP0, int ldP1,
                    const float* __restrict__ scaleMat, int scaleStride,
                    int K)
{
  __shared__ __align__(16) unsigned char aLds[TILE_M * LDSA_STRIDE];
  __shared__ __align__(16) unsigned char bLds[128    * LDSB_STRIDE];

  const float*          A   = (const float*)Aop;
  const unsigned short* Abf = (const unsigned short*)Aop;

  const unsigned short* Bt = (blockIdx.y == 0) ? Bt0 : Bt1;
  float*       C  = (blockIdx.y == 0) ? C0  : C1;
  const float* P  = (blockIdx.y == 0) ? P0  : P1;
  const int   ldP = (blockIdx.y == 0) ? ldP0 : ldP1;

  const int tid  = threadIdx.x;
  const int lane = tid & 31;
  const int wave = tid >> 5;
  const int mr   = lane & 15;
  const int half = lane >> 4;
  const long rowBase = (long)blockIdx.x * TILE_M;

  v8f acc[8];
#pragma unroll
  for (int j = 0; j < 8; ++j)
#pragma unroll
    for (int r = 0; r < 8; ++r) acc[j][r] = 0.0f;

  // A-fragment offsets (ISA 16-bit A layout): lanes 0-15 take K0-7 / K16-23,
  // lanes 16-31 take K8-15 / K24-31.
  const int aRow  = wave * 16 + mr;
  const int aOff0 = aRow * LDSA_STRIDE + half * 16;
  const int aOff1 = aOff0 + 32;

  const unsigned short* bRowPtr = Bt + (size_t)tid * ldBt;  // thread owns col n=tid

  for (int k0 = 0; k0 < K; k0 += TILE_K) {
    // -------- stage tiles (global, coalesced) --------
    float4 aStage[4];
    unsigned int aPackT[8];
    uint4 aDirect[2];
    if (AMODE == AM_BF16) {
#pragma unroll
      for (int i = 0; i < 2; ++i) {
        int s  = tid + i * BLK;               // 0..255
        int r  = s >> 2;                      // 0..63
        int c4 = s & 3;                       // 16B chunk within 64B row
        aDirect[i] = *(const uint4*)(Abf + (rowBase + r) * (long)ldA + k0 + c4 * 8);
      }
    } else if (!TRANSA) {
#pragma unroll
      for (int i = 0; i < 4; ++i) {
        int s  = tid + i * BLK;               // 0..511
        int r  = s >> 3;                      // 0..63
        int c4 = s & 7;
        aStage[i] = *(const float4*)(A + (rowBase + r) * (long)ldA + k0 + c4 * 4);
      }
    } else {
      int m     = tid & 63;
      int khalf = tid >> 6;
#pragma unroll
      for (int j = 0; j < 8; ++j) {
        int kk   = khalf * 16 + 2 * j;
        float lo = A[(long)(k0 + kk)     * ldA + rowBase + m];
        float hi = A[(long)(k0 + kk + 1) * ldA + rowBase + m];
        aPackT[j] = f2bf_pair(lo, hi);
      }
    }
    // B already bf16 + transposed: 64 contiguous bytes for this thread's column
    const uint4* bsrc = (const uint4*)(bRowPtr + k0);
    uint4 bq0 = bsrc[0], bq1 = bsrc[1], bq2 = bsrc[2], bq3 = bsrc[3];

    __syncthreads();   // previous iteration's LDS reads complete

    // -------- store to LDS (and optional bf16-A writeout) --------
    if (AMODE == AM_BF16) {
#pragma unroll
      for (int i = 0; i < 2; ++i) {
        int s  = tid + i * BLK;
        int r  = s >> 2;
        int c4 = s & 3;
        *(uint4*)(aLds + r * LDSA_STRIDE + c4 * 16) = aDirect[i];
      }
    } else if (!TRANSA) {
#pragma unroll
      for (int i = 0; i < 4; ++i) {
        int s  = tid + i * BLK;
        int r  = s >> 3;
        int c4 = s & 7;
        uint2 p;
        p.x = f2bf_pair(aStage[i].x, aStage[i].y);
        p.y = f2bf_pair(aStage[i].z, aStage[i].w);
        *(uint2*)(aLds + r * LDSA_STRIDE + c4 * 8) = p;
        if (AMODE == AM_WRITE && blockIdx.y == 0) {
          // persist bf16 copy (row stride == ldA == K for our callers)
          *(uint2*)(Aout + (rowBase + r) * (long)ldA + k0 + c4 * 4) = p;
        }
      }
    } else {
      int m     = tid & 63;
      int khalf = tid >> 6;
      *(uint4*)(aLds + m * LDSA_STRIDE + khalf * 32) =
          make_uint4(aPackT[0], aPackT[1], aPackT[2], aPackT[3]);
      *(uint4*)(aLds + m * LDSA_STRIDE + khalf * 32 + 16) =
          make_uint4(aPackT[4], aPackT[5], aPackT[6], aPackT[7]);
    }
    {
      unsigned char* row = bLds + tid * LDSB_STRIDE;   // [n][k]
      *(uint4*)(row +  0) = bq0;
      *(uint4*)(row + 16) = bq1;
      *(uint4*)(row + 32) = bq2;
      *(uint4*)(row + 48) = bq3;
    }
    __syncthreads();

    // -------- hoist all fragments, then 8 back-to-back WMMAs --------
    FragCvt af, bfr[8];
    af.u[0] = *(const uint4*)(aLds + aOff0);
    af.u[1] = *(const uint4*)(aLds + aOff1);
#pragma unroll
    for (int j = 0; j < 8; ++j) {
      const unsigned char* brow = bLds + (j * 16 + mr) * LDSB_STRIDE + half * 16;
      bfr[j].u[0] = *(const uint4*)(brow);
      bfr[j].u[1] = *(const uint4*)(brow + 32);
    }
#pragma unroll
    for (int j = 0; j < 8; ++j) {
      acc[j] = __builtin_amdgcn_wmma_f32_16x16x32_bf16(
                   false, af.v, false, bfr[j].v, (short)0, acc[j], false, false);
    }
  }

  // -------- epilogue (C/D layout: VGPR r -> M = r + half*8, N = mr) --------
  const long mBase = rowBase + wave * 16 + half * 8;
#pragma unroll
  for (int j = 0; j < 8; ++j) {
    int col = j * 16 + mr;
#pragma unroll
    for (int r = 0; r < 8; ++r) {
      long row = mBase + r;
      float v = acc[j][r];
      if (CHEB)     v = 2.0f * v - P[row * (long)ldP + col];
      if (ROWSCALE) v *= scaleMat[row * (long)scaleStride];
      if (RELU)     v = fmaxf(v, 0.0f);
      C[row * (long)ldC + col] = v;
    }
  }
}

// ---------------------------------------------------------------------------
// Helper kernels
// ---------------------------------------------------------------------------

// Bt[c][n] (bf16, row stride N) = src[n][c] * (dmat ? dmat[n*dstride] : 1)
__global__ __launch_bounds__(256)
void transpose_cvt_kernel(const float* __restrict__ src, int ldSrc,
                          const float* __restrict__ dmat, int dstride,
                          unsigned short* __restrict__ Bt, int N)
{
  __shared__ float tile[32][33];
  const int tx = threadIdx.x & 31;
  const int ty = threadIdx.x >> 5;               // 0..7
  const long nBase = (long)blockIdx.x * 32;
  const int  cBase = blockIdx.y * 32;            // 0,32,64,96
#pragma unroll
  for (int i = 0; i < 4; ++i) {
    int row = ty + i * 8;                        // local n
    float v = src[(nBase + row) * (long)ldSrc + cBase + tx];
    if (dmat) v *= dmat[(nBase + row) * (long)dstride];
    tile[row][tx] = v;
  }
  __syncthreads();
#pragma unroll
  for (int i = 0; i < 4; ++i) {
    int crow = ty + i * 8;                       // local c
    Bt[(long)(cBase + crow) * N + nBase + tx] = f2bf1(tile[tx][crow]);
  }
}

// X[n*LD + slotOff + f] = x[n,f]
__global__ void copy_slot_kernel(const float* __restrict__ x,
                                 float* __restrict__ X, int LD, int slotOff,
                                 int total) {
  int i = blockIdx.x * blockDim.x + threadIdx.x;
  if (i < total) {
    int n = i >> 7, f = i & 127;
    X[(long)n * LD + slotOff + f] = x[i];
  }
}

// Wt[o][k*128+ii] = bf16(w[(ii*128+o)*KT + k])
__global__ void repack_w_bt_kernel(const float* __restrict__ w,
                                   unsigned short* __restrict__ Wt,
                                   int KT, int total) {
  int i = blockIdx.x * blockDim.x + threadIdx.x;
  if (i < total) {
    int KR = KT * 128;
    int o  = i / KR;
    int r  = i - o * KR;       // k*128 + ii
    int k  = r >> 7;
    int ii = r & 127;
    Wt[i] = f2bf1(w[((long)(ii * 128 + o)) * KT + k]);
  }
}

// ---------------------------------------------------------------------------
// Orchestration
// ---------------------------------------------------------------------------

#define GEMM1(AM, TA, CH, RL, RS, Ap, lda, Ao, Btp, ldbt, Cp, ldc, Pp, ldp, Sp, ss, M, Kd) \
  gemm_bf16_wmma<AM, TA, CH, RL, RS><<<dim3((M) / TILE_M, 1), BLK, 0, stream>>>(           \
      Ap, lda, Ao, Btp, Btp, ldbt, Cp, Cp, ldc, Pp, Pp, ldp, ldp, Sp, ss, Kd)

#define GEMM2(AM, CH, Ap, lda, Ao, B0p, B1p, ldbt, C0p, C1p, ldc, P0p, P1p, lp0, lp1, M, Kd) \
  gemm_bf16_wmma<AM, false, CH, false, false><<<dim3((M) / TILE_M, 2), BLK, 0, stream>>>(    \
      Ap, lda, Ao, B0p, B1p, ldbt, C0p, C1p, ldc, P0p, P1p, lp0, lp1, nullptr, 0, Kd)

#define TCVT(srcp, lds_, dmatp, dstr, dstp, Nn) \
  transpose_cvt_kernel<<<dim3((Nn) / 32, 4), 256, 0, stream>>>(srcp, lds_, dmatp, dstr, dstp, Nn)

extern "C" void kernel_launch(void* const* d_in, const int* in_sizes, int n_in,
                              void* d_out, int out_size, void* d_ws, size_t ws_size,
                              hipStream_t stream) {
  (void)in_sizes; (void)n_in; (void)out_size;

  const float* x0      = (const float*)d_in[0];
  const float* x1      = (const float*)d_in[1];
  const float* x2      = (const float*)d_in[2];
  const float* b1      = (const float*)d_in[3];
  const float* b2      = (const float*)d_in[4];
  const float* l0      = (const float*)d_in[5];
  const float* l1l     = (const float*)d_in[6];
  const float* l1u     = (const float*)d_in[7];
  const float* l2      = (const float*)d_in[8];
  const float* d1_inv  = (const float*)d_in[9];
  const float* d3      = (const float*)d_in[10];
  const float* d5_pinv = (const float*)d_in[11];
  const float* w0      = (const float*)d_in[12];
  const float* w1      = (const float*)d_in[13];
  const float* w2      = (const float*)d_in[14];

  // ---- workspace carve-up ----
  const size_t f32Elems  = (size_t)N0 * 768 + (size_t)N1 * 1024 + (size_t)N2 * 768;
  const size_t bf16Elems = (size_t)128 * (4 * N0 + 6 * N1 + 5 * N2) + 128 * 2560;
  const size_t baseBytes = f32Elems * 4 + bf16Elems * 2;
  if (ws_size < baseBytes) return;
  // optional shared bf16-Laplacian scratch (reused by l0 -> l1l -> l1u -> l2)
  const size_t abfElems = (size_t)N1 * N1;             // 134 MB, largest (l1l/l1u)
  const bool useAbf = ws_size >= baseBytes + abfElems * 2 + 256;

  float* X0 = (float*)d_ws;                 // [2048 x 768]
  float* X1 = X0 + (size_t)N0 * 768;        // [8192 x 1024]
  float* X2 = X1 + (size_t)N1 * 1024;       // [4096 x 768]
  unsigned short* bp = (unsigned short*)(X2 + (size_t)N2 * 768);
  unsigned short* x0t  = bp; bp += (size_t)128 * N0;
  unsigned short* x1t  = bp; bp += (size_t)128 * N1;
  unsigned short* x2t  = bp; bp += (size_t)128 * N2;
  unsigned short* t1t  = bp; bp += (size_t)128 * N1;   // (d5_pinv*x1)^T
  unsigned short* t2t  = bp; bp += (size_t)128 * N2;   // (d3*x2)^T
  unsigned short* s01t = bp; bp += (size_t)128 * N0;
  unsigned short* s03t = bp; bp += (size_t)128 * N0;   // x0p ^T
  unsigned short* s04t = bp; bp += (size_t)128 * N0;
  unsigned short* s11t = bp; bp += (size_t)128 * N1;
  unsigned short* s13t = bp; bp += (size_t)128 * N1;
  unsigned short* s15t = bp; bp += (size_t)128 * N1;   // x1p ^T
  unsigned short* s16t = bp; bp += (size_t)128 * N1;
  unsigned short* s20t = bp; bp += (size_t)128 * N2;   // x2n ^T
  unsigned short* s21t = bp; bp += (size_t)128 * N2;
  unsigned short* s24t = bp; bp += (size_t)128 * N2;
  unsigned short* W0t  = bp; bp += (size_t)128 * 768;
  unsigned short* W1t  = bp; bp += (size_t)128 * 1024;
  unsigned short* W2t  = bp; bp += (size_t)128 * 768;
  unsigned short* Abf  = bp;                           // shared Laplacian bf16 copy
  float* out = (float*)d_out;
  const float* NP = nullptr;
  unsigned short* NS = nullptr;

  // ---- stage 0: raw-feature staging + bf16 transposed operands ----
  copy_slot_kernel<<<(N0 * 128) / 256, 256, 0, stream>>>(x0, X0, 768, 0, N0 * 128);
  copy_slot_kernel<<<(N1 * 128) / 256, 256, 0, stream>>>(x1, X1, 1024, 0, N1 * 128);
  copy_slot_kernel<<<(N2 * 128) / 256, 256, 0, stream>>>(x2, X2, 768, 3 * 128, N2 * 128);
  TCVT(x0, 128, NP, 0, x0t, N0);
  TCVT(x1, 128, NP, 0, x1t, N1);
  TCVT(x2, 128, NP, 0, x2t, N2);
  TCVT(x1, 128, d5_pinv, N1 + 1, t1t, N1);
  TCVT(x2, 128, d3,      N2 + 1, t2t, N2);
  repack_w_bt_kernel<<<(128 * 768)  / 256, 256, 0, stream>>>(w0, W0t, 6, 128 * 768);
  repack_w_bt_kernel<<<(128 * 1024) / 256, 256, 0, stream>>>(w1, W1t, 8, 128 * 1024);
  repack_w_bt_kernel<<<(128 * 768)  / 256, 256, 0, stream>>>(w2, W2t, 6, 128 * 768);

  // ---- order 0: X0 slots [x0, T1, T2, x0p, T1p, T2p] ----
  GEMM1(AM_F32, false, false, false, true, b1, N1, NS, x1t, N1,
        X0 + 3 * 128, 768, NP, 0, d1_inv, N0 + 1, N0, N1);      // x0p
  TCVT(X0 + 3 * 128, 768, NP, 0, s03t, N0);
  if (useAbf) {   // pass1 writes l0 bf16 copy; pass2 reads it (L2-resident)
    GEMM2(AM_WRITE, false, l0, N0, Abf, x0t, s03t, N0,
          X0 + 1 * 128, X0 + 4 * 128, 768, NP, NP, 0, 0, N0, N0);
    TCVT(X0 + 1 * 128, 768, NP, 0, s01t, N0);
    TCVT(X0 + 4 * 128, 768, NP, 0, s04t, N0);
    GEMM2(AM_BF16, true, Abf, N0, NS, s01t, s04t, N0,
          X0 + 2 * 128, X0 + 5 * 128, 768, x0, X0 + 3 * 128, 128, 768, N0, N0);
  } else {
    GEMM2(AM_F32, false, l0, N0, NS, x0t, s03t, N0,
          X0 + 1 * 128, X0 + 4 * 128, 768, NP, NP, 0, 0, N0, N0);
    TCVT(X0 + 1 * 128, 768, NP, 0, s01t, N0);
    TCVT(X0 + 4 * 128, 768, NP, 0, s04t, N0);
    GEMM2(AM_F32, true, l0, N0, NS, s01t, s04t, N0,
          X0 + 2 * 128, X0 + 5 * 128, 768, x0, X0 + 3 * 128, 128, 768, N0, N0);
  }

  // ---- order 1: X1 slots [x1, T1l, T2l, T1u, T2u, x1p, T1up, T2up] ----
  GEMM1(AM_F32, false, false, false, false, b2, N2, NS, t2t, N2,
        X1 + 5 * 128, 1024, NP, 0, NP, 0, N1, N2);              // x1p
  TCVT(X1 + 5 * 128, 1024, NP, 0, s15t, N1);
  if (useAbf) {
    // l1l chain (sequential), Abf reused
    GEMM1(AM_WRITE, false, false, false, false, l1l, N1, Abf, x1t, N1,
          X1 + 1 * 128, 1024, NP, 0, NP, 0, N1, N1);
    TCVT(X1 + 1 * 128, 1024, NP, 0, s11t, N1);
    GEMM1(AM_BF16, false, true, false, false, Abf, N1, NS, s11t, N1,
          X1 + 2 * 128, 1024, x1, 128, NP, 0, N1, N1);
    // l1u fused chains, Abf reused
    GEMM2(AM_WRITE, false, l1u, N1, Abf, x1t, s15t, N1,
          X1 + 3 * 128, X1 + 6 * 128, 1024, NP, NP, 0, 0, N1, N1);
    TCVT(X1 + 3 * 128, 1024, NP, 0, s13t, N1);
    TCVT(X1 + 6 * 128, 1024, NP, 0, s16t, N1);
    GEMM2(AM_BF16, true, Abf, N1, NS, s13t, s16t, N1,
          X1 + 4 * 128, X1 + 7 * 128, 1024, x1, X1 + 5 * 128, 128, 1024, N1, N1);
  } else {
    GEMM1(AM_F32, false, false, false, false, l1l, N1, NS, x1t, N1,
          X1 + 1 * 128, 1024, NP, 0, NP, 0, N1, N1);
    TCVT(X1 + 1 * 128, 1024, NP, 0, s11t, N1);
    GEMM1(AM_F32, false, true, false, false, l1l, N1, NS, s11t, N1,
          X1 + 2 * 128, 1024, x1, 128, NP, 0, N1, N1);
    GEMM2(AM_F32, false, l1u, N1, NS, x1t, s15t, N1,
          X1 + 3 * 128, X1 + 6 * 128, 1024, NP, NP, 0, 0, N1, N1);
    TCVT(X1 + 3 * 128, 1024, NP, 0, s13t, N1);
    TCVT(X1 + 6 * 128, 1024, NP, 0, s16t, N1);
    GEMM2(AM_F32, true, l1u, N1, NS, s13t, s16t, N1,
          X1 + 4 * 128, X1 + 7 * 128, 1024, x1, X1 + 5 * 128, 128, 1024, N1, N1);
  }

  // ---- order 2: X2 slots [x2n, T1n, T2n, x2, T1, T2] ----
  GEMM1(AM_F32, true, false, false, false, b2, N2, NS, t1t, N1,
        X2 + 0 * 128, 768, NP, 0, NP, 0, N2, N1);               // x2n = b2^T @ ...
  TCVT(X2 + 0 * 128, 768, NP, 0, s20t, N2);
  if (useAbf) {
    GEMM2(AM_WRITE, false, l2, N2, Abf, s20t, x2t, N2,
          X2 + 1 * 128, X2 + 4 * 128, 768, NP, NP, 0, 0, N2, N2);
    TCVT(X2 + 1 * 128, 768, NP, 0, s21t, N2);
    TCVT(X2 + 4 * 128, 768, NP, 0, s24t, N2);
    GEMM2(AM_BF16, true, Abf, N2, NS, s21t, s24t, N2,
          X2 + 2 * 128, X2 + 5 * 128, 768, X2 + 0 * 128, x2, 768, 128, N2, N2);
  } else {
    GEMM2(AM_F32, false, l2, N2, NS, s20t, x2t, N2,
          X2 + 1 * 128, X2 + 4 * 128, 768, NP, NP, 0, 0, N2, N2);
    TCVT(X2 + 1 * 128, 768, NP, 0, s21t, N2);
    TCVT(X2 + 4 * 128, 768, NP, 0, s24t, N2);
    GEMM2(AM_F32, true, l2, N2, NS, s21t, s24t, N2,
          X2 + 2 * 128, X2 + 5 * 128, 768, X2 + 0 * 128, x2, 768, 128, N2, N2);
  }

  // ---- final: y = relu(Xcat @ W), outputs concatenated (y0, y1, y2) ----
  GEMM1(AM_F32, false, false, true, false, X0, 768, NS, W0t, 768,
        out, 128, NP, 0, NP, 0, N0, 768);
  GEMM1(AM_F32, false, false, true, false, X1, 1024, NS, W1t, 1024,
        out + (size_t)N0 * 128, 128, NP, 0, NP, 0, N1, 1024);
  GEMM1(AM_F32, false, false, true, false, X2, 768, NS, W2t, 768,
        out + (size_t)(N0 + N1) * 128, 128, NP, 0, NP, 0, N2, 768);
}
